// CurveSOTAQueryNet_60361470378236
// MI455X (gfx1250) — compile-verified, matrix operand order checked by reading, not run on hardware
//
#include <hip/hip_runtime.h>

// ---------------------------------------------------------------------------
// Types for CDNA5 WMMA (wave32, 16x16x32 bf16 -> f32)
// ---------------------------------------------------------------------------
typedef __attribute__((ext_vector_type(16))) __bf16 v16bf;
typedef __attribute__((ext_vector_type(8)))  __bf16 v8bf;
typedef __attribute__((ext_vector_type(8)))  float  v8f;

__device__ __forceinline__ __bf16 f2bf(float f) {
  unsigned u = __float_as_uint(f);
  unsigned r = u + 0x7FFFu + ((u >> 16) & 1u);   // round to nearest even
  unsigned short h = (unsigned short)(r >> 16);
  return __builtin_bit_cast(__bf16, h);
}
__device__ __forceinline__ float bf2f(__bf16 h) {
  unsigned u = ((unsigned)__builtin_bit_cast(unsigned short, h)) << 16;
  return __uint_as_float(u);
}
// pack two fp32 -> two bf16 (RNE) in one dword
__device__ __forceinline__ unsigned packbf2(float a, float b) {
  unsigned ua = __float_as_uint(a); ua = ua + 0x7FFFu + ((ua >> 16) & 1u);
  unsigned ub = __float_as_uint(b); ub = ub + 0x7FFFu + ((ub >> 16) & 1u);
  return (ua >> 16) | (ub & 0xFFFF0000u);
}

// ---------------------------------------------------------------------------
// Generic batched GEMM:  C[b1,b2] = alpha * A[b1,b2] @ op(B[b1,b2]) + bias
//   TRANS_B==1 : B is (N,K) row-major ; TRANS_B==0 : B is (K,N) row-major
//   IN_BF16==1 : A and B are packed bf16 in global memory (no convert on load)
//   OUT_BF16==1: C is stored as packed bf16
// Block tile 128x128, K-step 32, 256 threads = 8 waves, each wave 32x64.
// ---------------------------------------------------------------------------
#define BM 128
#define BN 128
#define BKs 32
#define LDT 40   // LDS row stride in bf16 elems (80B -> conflict-free B128 reads)

template <int TRANS_B, int IN_BF16, int OUT_BF16>
__global__ void __launch_bounds__(256)
gemm_wmma(const void* __restrict__ Av, const void* __restrict__ Bv,
          const float* __restrict__ bias, void* __restrict__ Cv,
          int M, int N, int K, int lda, int ldb, int ldc,
          int batch2,
          long long sa1, long long sb1, long long sc1,
          long long sa2, long long sb2, long long sc2,
          float alpha)
{
  __shared__ __align__(16) __bf16 sA[BM * LDT];
  __shared__ __align__(16) __bf16 sB[BN * LDT];

  const int t  = threadIdx.x;
  const int b2 = (int)(blockIdx.z % (unsigned)batch2);
  const int b1 = (int)(blockIdx.z / (unsigned)batch2);
  const long long aoff = (long long)b1 * sa1 + (long long)b2 * sa2;
  const long long boff = (long long)b1 * sb1 + (long long)b2 * sb2;
  const long long coff = (long long)b1 * sc1 + (long long)b2 * sc2;
  const float*  Af  = (const float*)Av  + aoff;
  const __bf16* Ah  = (const __bf16*)Av + aoff;
  const float*  Bf  = (const float*)Bv  + boff;
  const __bf16* Bh  = (const __bf16*)Bv + boff;

  const int n0 = blockIdx.x * BN;
  const int m0 = blockIdx.y * BM;

  const int wid   = t >> 5;
  const int lane  = t & 31;
  const int wm    = (wid & 3) * 32;   // wave row offset within tile
  const int wn    = (wid >> 2) * 64;  // wave col offset within tile
  const int lrow  = lane & 15;
  const int khalf = (lane >> 4) * 8;

  const bool fullA = (m0 + BM <= M);
  const bool fullB = (n0 + BN <= N);

  // staging coordinates
  const int srow = t >> 3;            // f32 path: row base (0..31), step 32
  const int skq  = (t & 7) * 4;       // f32 path: k offset 0..28
  const int hrow = t >> 2;            // bf16 path: row base (0..63), step 64
  const int hkq  = (t & 3) * 8;       // bf16 path: k offset 0,8,16,24

  v8f acc[2][4];
#pragma unroll
  for (int i = 0; i < 2; ++i)
#pragma unroll
    for (int j = 0; j < 4; ++j)
      acc[i][j] = (v8f){0.f,0.f,0.f,0.f,0.f,0.f,0.f,0.f};

  for (int k0 = 0; k0 < K; k0 += BKs) {
    const bool kfull = (k0 + BKs <= K);

    // ---- stage A tile (BM x 32)
    if (IN_BF16) {
      if (fullA && kfull) {
#pragma unroll
        for (int it = 0; it < 2; ++it) {
          int r = hrow + it * 64;
          const __bf16* gp = Ah + (size_t)(m0 + r) * lda + k0 + hkq;
          if (k0 + BKs < K) __builtin_prefetch((const char*)(gp + BKs), 0, 3);
          *(v8bf*)(&sA[r * LDT + hkq]) = *(const v8bf*)gp;
        }
      } else {
#pragma unroll
        for (int it = 0; it < 4; ++it) {
          int r = srow + it * 32;
          int gm = m0 + r;
#pragma unroll
          for (int j = 0; j < 4; ++j) {
            int gk = k0 + skq + j;
            sA[r * LDT + skq + j] = (gm < M && gk < K)
                ? Ah[(size_t)gm * lda + gk]
                : __builtin_bit_cast(__bf16, (unsigned short)0);
          }
        }
      }
    } else {
      if (fullA && kfull) {
#pragma unroll
        for (int it = 0; it < 4; ++it) {
          int r = srow + it * 32;
          const float* gp = Af + (size_t)(m0 + r) * lda + k0 + skq;
          if (it == 0 && k0 + BKs < K) __builtin_prefetch((const char*)(gp + BKs), 0, 3);
          const float4 v = *(const float4*)gp;
          uint2 p; p.x = packbf2(v.x, v.y); p.y = packbf2(v.z, v.w);
          *(uint2*)(&sA[r * LDT + skq]) = p;
        }
      } else {
#pragma unroll
        for (int it = 0; it < 4; ++it) {
          int r = srow + it * 32;
          int gm = m0 + r;
#pragma unroll
          for (int j = 0; j < 4; ++j) {
            int gk = k0 + skq + j;
            float v = (gm < M && gk < K) ? Af[(size_t)gm * lda + gk] : 0.f;
            sA[r * LDT + skq + j] = f2bf(v);
          }
        }
      }
    }
    // ---- stage B tile
    if (TRANS_B) {                   // B rows are the N dimension
      if (IN_BF16) {
        if (fullB && kfull) {
#pragma unroll
          for (int it = 0; it < 2; ++it) {
            int r = hrow + it * 64;
            const __bf16* gp = Bh + (size_t)(n0 + r) * ldb + k0 + hkq;
            if (k0 + BKs < K) __builtin_prefetch((const char*)(gp + BKs), 0, 3);
            *(v8bf*)(&sB[r * LDT + hkq]) = *(const v8bf*)gp;
          }
        } else {
#pragma unroll
          for (int it = 0; it < 4; ++it) {
            int r = srow + it * 32;
            int gn = n0 + r;
#pragma unroll
            for (int j = 0; j < 4; ++j) {
              int gk = k0 + skq + j;
              sB[r * LDT + skq + j] = (gn < N && gk < K)
                  ? Bh[(size_t)gn * ldb + gk]
                  : __builtin_bit_cast(__bf16, (unsigned short)0);
            }
          }
        }
      } else {
        if (fullB && kfull) {
#pragma unroll
          for (int it = 0; it < 4; ++it) {
            int r = srow + it * 32;
            const float* gp = Bf + (size_t)(n0 + r) * ldb + k0 + skq;
            if (it == 0 && k0 + BKs < K) __builtin_prefetch((const char*)(gp + BKs), 0, 3);
            const float4 v = *(const float4*)gp;
            uint2 p; p.x = packbf2(v.x, v.y); p.y = packbf2(v.z, v.w);
            *(uint2*)(&sB[r * LDT + skq]) = p;
          }
        } else {
#pragma unroll
          for (int it = 0; it < 4; ++it) {
            int r = srow + it * 32;
            int gn = n0 + r;
#pragma unroll
            for (int j = 0; j < 4; ++j) {
              int gk = k0 + skq + j;
              float v = (gn < N && gk < K) ? Bf[(size_t)gn * ldb + gk] : 0.f;
              sB[r * LDT + skq + j] = f2bf(v);
            }
          }
        }
      }
    } else {                          // B is (K,N): stage transposed into LDS (f32 only)
      const int kr = t >> 5;          // k row 0..7 (step 8)
      const int nq = (t & 31) * 4;    // n 0..124
      if (fullB && kfull) {
#pragma unroll
        for (int it = 0; it < 4; ++it) {
          int gk = k0 + kr + it * 8;
          const float4 v = *(const float4*)(Bf + (size_t)gk * ldb + n0 + nq);
          int kl = kr + it * 8;
          sB[(nq + 0) * LDT + kl] = f2bf(v.x);
          sB[(nq + 1) * LDT + kl] = f2bf(v.y);
          sB[(nq + 2) * LDT + kl] = f2bf(v.z);
          sB[(nq + 3) * LDT + kl] = f2bf(v.w);
        }
      } else {
#pragma unroll
        for (int it = 0; it < 4; ++it) {
          int kl = kr + it * 8;
          int gk = k0 + kl;
#pragma unroll
          for (int j = 0; j < 4; ++j) {
            int gn = n0 + nq + j;
            float v = (gk < K && gn < N) ? Bf[(size_t)gk * ldb + gn] : 0.f;
            sB[(nq + j) * LDT + kl] = f2bf(v);
          }
        }
      }
    }
    __syncthreads();

    // ---- fragments per ISA layout: lane L -> row L%16, K halves {kh..kh+7, 16+kh..}
    v16bf afrag[2], bfrag[4];
#pragma unroll
    for (int i = 0; i < 2; ++i) {
      const __bf16* p = &sA[(wm + i * 16 + lrow) * LDT + khalf];
      v8bf lo = *(const v8bf*)p;
      v8bf hi = *(const v8bf*)(p + 16);
      afrag[i] = __builtin_shufflevector(lo, hi, 0,1,2,3,4,5,6,7,8,9,10,11,12,13,14,15);
    }
#pragma unroll
    for (int j = 0; j < 4; ++j) {
      const __bf16* p = &sB[(wn + j * 16 + lrow) * LDT + khalf];
      v8bf lo = *(const v8bf*)p;
      v8bf hi = *(const v8bf*)(p + 16);
      bfrag[j] = __builtin_shufflevector(lo, hi, 0,1,2,3,4,5,6,7,8,9,10,11,12,13,14,15);
    }
#pragma unroll
    for (int i = 0; i < 2; ++i)
#pragma unroll
      for (int j = 0; j < 4; ++j)
        acc[i][j] = __builtin_amdgcn_wmma_f32_16x16x32_bf16(
            false, afrag[i], false, bfrag[j], (short)0, acc[i][j], false, false);
    __syncthreads();
  }

  // ---- store: VGPR r -> M = r + 8*(lane>>4); N = lane&15
  const int cn   = lane & 15;
  const int rofs = (lane >> 4) * 8;
#pragma unroll
  for (int i = 0; i < 2; ++i) {
#pragma unroll
    for (int j = 0; j < 4; ++j) {
      int gcol = n0 + wn + j * 16 + cn;
      if (gcol >= N) continue;
      float bb = bias ? bias[gcol] : 0.f;
#pragma unroll
      for (int r = 0; r < 8; ++r) {
        int grow = m0 + wm + i * 16 + rofs + r;
        if (grow < M) {
          float v = acc[i][j][r] * alpha + bb;
          if (OUT_BF16)
            ((__bf16*)Cv + coff)[(size_t)grow * ldc + gcol] = f2bf(v);
          else
            ((float*)Cv + coff)[(size_t)grow * ldc + gcol] = v;
        }
      }
    }
  }
}

// ---------------------------------------------------------------------------
// Elementwise / reduction kernels
// ---------------------------------------------------------------------------
__global__ void l2norm_rows_bf16(__bf16* __restrict__ x, int D) {
  __bf16* row = x + (size_t)blockIdx.x * D;
  int t = threadIdx.x;                 // blockDim == D == 256
  float v = bf2f(row[t]);
  __shared__ float red[256];
  red[t] = v * v; __syncthreads();
  for (int s = 128; s > 0; s >>= 1) { if (t < s) red[t] += red[t + s]; __syncthreads(); }
  float inv = 1.f / fmaxf(sqrtf(red[0]), 1e-6f);
  row[t] = f2bf(v * inv);
}

__global__ void softmax_rows(float* __restrict__ x, int L) {
  float* row = x + (size_t)blockIdx.x * L;
  int t = threadIdx.x;
  __shared__ float red[256];
  float m = -INFINITY;
  for (int i = t; i < L; i += 256) m = fmaxf(m, row[i]);
  red[t] = m; __syncthreads();
  for (int s = 128; s > 0; s >>= 1) { if (t < s) red[t] = fmaxf(red[t], red[t + s]); __syncthreads(); }
  m = red[0]; __syncthreads();
  float sum = 0.f;
  for (int i = t; i < L; i += 256) { float e = expf(row[i] - m); row[i] = e; sum += e; }
  red[t] = sum; __syncthreads();
  for (int s = 128; s > 0; s >>= 1) { if (t < s) red[t] += red[t + s]; __syncthreads(); }
  float inv = 1.f / red[0];
  for (int i = t; i < L; i += 256) row[i] *= inv;
}

__global__ void residual_ln(float* __restrict__ out, const float* __restrict__ x,
                            const float* __restrict__ y, const float* __restrict__ w,
                            const float* __restrict__ b, int D) {
  size_t r = blockIdx.x; int t = threadIdx.x;     // blockDim == D == 256
  float v = x[r * D + t] + y[r * D + t];
  __shared__ float red[256];
  red[t] = v; __syncthreads();
  for (int s = 128; s > 0; s >>= 1) { if (t < s) red[t] += red[t + s]; __syncthreads(); }
  float mean = red[0] / (float)D; __syncthreads();
  float d = v - mean;
  red[t] = d * d; __syncthreads();
  for (int s = 128; s > 0; s >>= 1) { if (t < s) red[t] += red[t + s]; __syncthreads(); }
  float var = red[0] / (float)D;
  out[r * D + t] = d * rsqrtf(var + 1e-5f) * w[t] + b[t];
}

__global__ void gelu_inplace(float* __restrict__ x, size_t n) {
  size_t i = (size_t)blockIdx.x * 256 + threadIdx.x;
  if (i < n) { float v = x[i]; x[i] = 0.5f * v * (1.f + erff(v * 0.70710678118654752f)); }
}

__global__ void addv(float* __restrict__ o, const float* __restrict__ a,
                     const float* __restrict__ b, size_t n) {
  size_t i = (size_t)blockIdx.x * 256 + threadIdx.x;
  if (i < n) o[i] = a[i] + b[i];
}

__global__ void gate_combine(float* __restrict__ qo, const float* __restrict__ qi,
                             const float* __restrict__ g2, const float* __restrict__ al,
                             size_t n) {
  size_t i = (size_t)blockIdx.x * 256 + threadIdx.x;
  if (i < n) qo[i] = qi[i] + al[i] / (1.f + expf(-g2[i]));
}

__global__ void concat2(const float* __restrict__ a, const float* __restrict__ b,
                        float* __restrict__ o, size_t rows, int D) {
  size_t i = (size_t)blockIdx.x * 256 + threadIdx.x;
  size_t total = rows * (size_t)(2 * D);
  if (i >= total) return;
  int c = (int)(i % (2 * D)); size_t r = i / (2 * D);
  o[i] = (c < D) ? a[r * D + c] : b[r * D + (c - D)];
}

__global__ void colmax(const float* __restrict__ sim, float* __restrict__ imp, int Qn, int N) {
  int b = blockIdx.y; int n = blockIdx.x * 256 + threadIdx.x;
  if (n >= N) return;
  const float* p = sim + (size_t)b * Qn * N + n;
  float m = -INFINITY;
  for (int q = 0; q < Qn; ++q) m = fmaxf(m, p[(size_t)q * N]);
  imp[(size_t)b * N + n] = m;
}

__global__ void gather_rows(const float* __restrict__ mem, const int* __restrict__ idx,
                            float* __restrict__ out, int N, int K, int D) {
  int bk = blockIdx.x; int b = bk / K; int k = bk % K; int t = threadIdx.x;
  out[(size_t)bk * D + t] = mem[((size_t)b * N + idx[(size_t)b * K + k]) * D + t];
}

// ---------------------------------------------------------------------------
// Block-level radix top-K (exactly K indices, order-free; 3 passes over row)
// ---------------------------------------------------------------------------
__device__ __forceinline__ unsigned fkey(float f) {
  unsigned u = __float_as_uint(f);
  return (u & 0x80000000u) ? ~u : (u | 0x80000000u);   // monotone order key
}

__global__ void topk_radix(const float* __restrict__ src, int N, int K,
                           int* __restrict__ idx_out, float* __restrict__ val_out) {
  const float* row = src + (size_t)blockIdx.x * N;
  __shared__ int hist[256];
  __shared__ unsigned sh_t1, sh_t2;
  __shared__ int sh_base, sh_strict, cstrict, cequal;
  int t = threadIdx.x;

  hist[t] = 0; __syncthreads();
  for (int i = t; i < N; i += 256) atomicAdd(&hist[fkey(row[i]) >> 24], 1);
  __syncthreads();
  if (t == 0) {
    int acc = 0, b;
    for (b = 255; b >= 0; --b) { if (acc + hist[b] >= K) break; acc += hist[b]; }
    sh_t1 = (unsigned)b; sh_base = acc;
  }
  __syncthreads();
  unsigned t1 = sh_t1;
  hist[t] = 0; __syncthreads();
  for (int i = t; i < N; i += 256) {
    unsigned k = fkey(row[i]);
    if ((k >> 24) == t1) atomicAdd(&hist[(k >> 16) & 0xFF], 1);
  }
  __syncthreads();
  if (t == 0) {
    int acc = sh_base; int b;
    for (b = 255; b >= 0; --b) { if (acc + hist[b] >= K) break; acc += hist[b]; }
    sh_t2 = (unsigned)b; sh_strict = acc; cstrict = 0; cequal = 0;
  }
  __syncthreads();
  unsigned thresh = (t1 << 8) | sh_t2;
  int strict = sh_strict;
  size_t base = (size_t)blockIdx.x * K;
  for (int i = t; i < N; i += 256) {
    float v = row[i];
    unsigned k16 = fkey(v) >> 16;
    if (k16 > thresh) {
      int p = atomicAdd(&cstrict, 1);
      idx_out[base + p] = i;
      if (val_out) val_out[base + p] = v;
    } else if (k16 == thresh) {
      int p = atomicAdd(&cequal, 1);
      if (strict + p < K) {
        idx_out[base + strict + p] = i;
        if (val_out) val_out[base + strict + p] = v;
      }
    }
  }
}

// softmax over K=96 vals then weighted gather-sum of memory rows
__global__ void align_agg(const float* __restrict__ mem, const int* __restrict__ idx,
                          const float* __restrict__ vals, float* __restrict__ out,
                          int Qn, int N, int K, int D) {
  int bq = blockIdx.x; int b = bq / Qn; int t = threadIdx.x;  // blockDim == D
  __shared__ float w[96];
  __shared__ int sidx[96];
  if (t < K) { w[t] = vals[(size_t)bq * K + t]; sidx[t] = idx[(size_t)bq * K + t]; }
  __syncthreads();
  if (t == 0) {
    float m = -INFINITY;
    for (int k = 0; k < K; ++k) m = fmaxf(m, w[k]);
    float s = 0.f;
    for (int k = 0; k < K; ++k) { w[k] = expf(w[k] - m); s += w[k]; }
    float inv = 1.f / s;
    for (int k = 0; k < K; ++k) w[k] *= inv;
  }
  __syncthreads();
  const float* mb = mem + (size_t)b * N * D;
  float acc = 0.f;
  for (int k = 0; k < K; ++k) acc += w[k] * mb[(size_t)sidx[k] * D + t];
  out[(size_t)bq * D + t] = acc;
}

// ---------------------------------------------------------------------------
// Host orchestration
// ---------------------------------------------------------------------------
static void gemm(hipStream_t s, bool tb, bool inbf, bool outbf,
                 const void* A, const void* B, const float* bias, void* C,
                 int M, int N, int K, int lda, int ldb, int ldc,
                 int b1 = 1, int b2 = 1,
                 long long sa1 = 0, long long sb1 = 0, long long sc1 = 0,
                 long long sa2 = 0, long long sb2 = 0, long long sc2 = 0,
                 float alpha = 1.f) {
  dim3 g((N + BN - 1) / BN, (M + BM - 1) / BM, b1 * b2);
  if (!tb)
    gemm_wmma<0,0,0><<<g, 256, 0, s>>>(A, B, bias, C, M, N, K, lda, ldb, ldc,
                                       b2, sa1, sb1, sc1, sa2, sb2, sc2, alpha);
  else if (inbf)
    gemm_wmma<1,1,0><<<g, 256, 0, s>>>(A, B, bias, C, M, N, K, lda, ldb, ldc,
                                       b2, sa1, sb1, sc1, sa2, sb2, sc2, alpha);
  else if (outbf)
    gemm_wmma<1,0,1><<<g, 256, 0, s>>>(A, B, bias, C, M, N, K, lda, ldb, ldc,
                                       b2, sa1, sb1, sc1, sa2, sb2, sc2, alpha);
  else
    gemm_wmma<1,0,0><<<g, 256, 0, s>>>(A, B, bias, C, M, N, K, lda, ldb, ldc,
                                       b2, sa1, sb1, sc1, sa2, sb2, sc2, alpha);
}

extern "C" void kernel_launch(void* const* d_in, const int* in_sizes, int n_in,
                              void* d_out, int out_size, void* d_ws, size_t ws_size,
                              hipStream_t stream) {
  (void)in_sizes; (void)n_in; (void)out_size; (void)ws_size;
  const int Bn = 8, Qn = 300, Nn = 16384, D = 256, H = 8, AK = 96, CK = 1024, FFn = 1024;
  const int BQ = Bn * Qn;                 // 2400
  const size_t QD = (size_t)BQ * D;       // 614400
  const float scale = 0.17677669529663687f;  // 1/sqrt(32)

  const float* q      = (const float*)d_in[0];
  const float* q_pos  = (const float*)d_in[1];
  const float* memory = (const float*)d_in[2];
  const float* Waq    = (const float*)d_in[3];
  const float* Wam    = (const float*)d_in[4];
  const float* Wg1    = (const float*)d_in[5];
  const float* bg1    = (const float*)d_in[6];
  const float* Wg2    = (const float*)d_in[7];
  const float* bg2    = (const float*)d_in[8];
  const float* Wcq    = (const float*)d_in[9];
  const float* Wcm    = (const float*)d_in[10];
  const float* saWqkv = (const float*)d_in[11];
  const float* sabqkv = (const float*)d_in[12];
  const float* saWo   = (const float*)d_in[13];
  const float* sabo   = (const float*)d_in[14];
  const float* caWqkv = (const float*)d_in[15];
  const float* cabqkv = (const float*)d_in[16];
  const float* caWo   = (const float*)d_in[17];
  const float* cabo   = (const float*)d_in[18];
  const float* Wff1   = (const float*)d_in[19];
  const float* bff1   = (const float*)d_in[20];
  const float* Wff2   = (const float*)d_in[21];
  const float* bff2   = (const float*)d_in[22];
  const float* ln1w = (const float*)d_in[23]; const float* ln1b = (const float*)d_in[24];
  const float* ln2w = (const float*)d_in[25]; const float* ln2b = (const float*)d_in[26];
  const float* ln3w = (const float*)d_in[27]; const float* ln3b = (const float*)d_in[28];

  // workspace arena (byte-based)
  char* wsp = (char*)d_ws; size_t off = 0;
  auto alloc = [&](size_t bytes) -> void* {
    void* p = (void*)(wsp + off);
    off += ((bytes + 255) & ~(size_t)255);
    return p;
  };
  __bf16* mn     = (__bf16*)alloc((size_t)Bn * Nn * D * 2);   // 64 MB bf16 (mn / mn2)
  float* sim     = (float*)alloc((size_t)BQ * Nn * 4);        // 157 MB (sim / sim2 / cscores)
  __bf16* qn     = (__bf16*)alloc((size_t)QD * 2);            // bf16
  int*   aidx    = (int*)alloc((size_t)BQ * AK * 4);
  float* avals   = (float*)alloc((size_t)BQ * AK * 4);
  float* aligned = (float*)alloc(QD * 4);
  float* cat     = (float*)alloc((size_t)BQ * 2 * D * 4);
  float* g1      = (float*)alloc(QD * 4);
  float* g2      = (float*)alloc(QD * 4);
  float* qbuf    = (float*)alloc(QD * 4);
  float* qk      = (float*)alloc(QD * 4);
  float* Qp      = (float*)alloc(QD * 4);
  float* Kp      = (float*)alloc((size_t)Bn * CK * D * 4);    // self & cross K
  float* Vp      = (float*)alloc((size_t)Bn * CK * D * 4);
  float* scores  = (float*)alloc((size_t)Bn * H * Qn * Qn * 4); // 23 MB
  float* ao      = (float*)alloc(QD * 4);
  float* proj    = (float*)alloc(QD * 4);
  float* imp     = (float*)alloc((size_t)Bn * Nn * 4);
  int*   idx2    = (int*)alloc((size_t)Bn * CK * 4);
  float* memsel  = (float*)alloc((size_t)Bn * CK * D * 4);
  float* f1      = (float*)alloc((size_t)BQ * FFn * 4);
  float* csc     = sim;                         // reuse sim region for cross scores

  // ---- 1. alignment: qn = l2norm(q@Waq.T) [bf16], mn = l2norm(memory@Wam.T) [bf16]
  gemm(stream, true, false, true, q, Waq, nullptr, qn, BQ, D, D, D, D, D);
  l2norm_rows_bf16<<<BQ, 256, 0, stream>>>(qn, D);
  gemm(stream, true, false, true, memory, Wam, nullptr, mn, Bn * Nn, D, D, D, D, D);
  l2norm_rows_bf16<<<Bn * Nn, 256, 0, stream>>>(mn, D);
  // sim[b] = qn[b] @ mn[b].T   (batched NT, bf16 operands)
  gemm(stream, true, true, false, qn, mn, nullptr, sim, Qn, Nn, D, D, D, Nn,
       Bn, 1, (long long)Qn * D, (long long)Nn * D, (long long)Qn * Nn);
  topk_radix<<<BQ, 256, 0, stream>>>(sim, Nn, AK, aidx, avals);
  align_agg<<<BQ, 256, 0, stream>>>(memory, aidx, avals, aligned, Qn, Nn, AK, D);
  // gate
  concat2<<<(unsigned)((QD * 2 + 255) / 256), 256, 0, stream>>>(q, aligned, cat, BQ, D);
  gemm(stream, true, false, false, cat, Wg1, bg1, g1, BQ, D, 2 * D, 2 * D, 2 * D, D);
  gelu_inplace<<<(unsigned)((QD + 255) / 256), 256, 0, stream>>>(g1, QD);
  gemm(stream, true, false, false, g1, Wg2, bg2, g2, BQ, D, D, D, D, D);
  gate_combine<<<(unsigned)((QD + 255) / 256), 256, 0, stream>>>(qbuf, q, g2, aligned, QD);

  // ---- 2. self attention (+ln1)
  addv<<<(unsigned)((QD + 255) / 256), 256, 0, stream>>>(qk, qbuf, q_pos, QD);
  gemm(stream, true, false, false, qk,   saWqkv,             sabqkv,       Qp, BQ, D, D, D, D, D);
  gemm(stream, true, false, false, qk,   saWqkv + 256 * 256, sabqkv + 256, Kp, BQ, D, D, D, D, D);
  gemm(stream, true, false, false, qbuf, saWqkv + 512 * 256, sabqkv + 512, Vp, BQ, D, D, D, D, D);
  // scores[b,h] = Qp[b,:,h*32:] @ Kp[b,:,h*32:].T * scale
  gemm(stream, true, false, false, Qp, Kp, nullptr, scores, Qn, Qn, 32, D, D, Qn,
       Bn, H, (long long)Qn * D, (long long)Qn * D, (long long)H * Qn * Qn,
       32, 32, (long long)Qn * Qn, scale);
  softmax_rows<<<Bn * H * Qn, 256, 0, stream>>>(scores, Qn);
  // ao[b,:,h*32:] = attn[b,h] @ Vp[b,:,h*32:]   (NN)
  gemm(stream, false, false, false, scores, Vp, nullptr, ao, Qn, 32, Qn, Qn, D, D,
       Bn, H, (long long)H * Qn * Qn, (long long)Qn * D, (long long)Qn * D,
       (long long)Qn * Qn, 32, 32, 1.f);
  gemm(stream, true, false, false, ao, saWo, sabo, proj, BQ, D, D, D, D, D);
  residual_ln<<<BQ, 256, 0, stream>>>(qbuf, qbuf, proj, ln1w, ln1b, D);

  // ---- 3. cross-memory sparsification
  gemm(stream, true, false, true, qbuf, Wcq, nullptr, qn, BQ, D, D, D, D, D);
  l2norm_rows_bf16<<<BQ, 256, 0, stream>>>(qn, D);
  gemm(stream, true, false, true, memory, Wcm, nullptr, mn, Bn * Nn, D, D, D, D, D);
  l2norm_rows_bf16<<<Bn * Nn, 256, 0, stream>>>(mn, D);
  gemm(stream, true, true, false, qn, mn, nullptr, sim, Qn, Nn, D, D, D, Nn,
       Bn, 1, (long long)Qn * D, (long long)Nn * D, (long long)Qn * Nn);
  colmax<<<dim3((Nn + 255) / 256, Bn), 256, 0, stream>>>(sim, imp, Qn, Nn);
  topk_radix<<<Bn, 256, 0, stream>>>(imp, Nn, CK, idx2, nullptr);
  gather_rows<<<Bn * CK, 256, 0, stream>>>(memory, idx2, memsel, Nn, CK, D);

  // ---- 4. cross attention (+ln2)
  addv<<<(unsigned)((QD + 255) / 256), 256, 0, stream>>>(qk, qbuf, q_pos, QD);
  gemm(stream, true, false, false, qk,     caWqkv,             cabqkv,       Qp, BQ,      D, D, D, D, D);
  gemm(stream, true, false, false, memsel, caWqkv + 256 * 256, cabqkv + 256, Kp, Bn * CK, D, D, D, D, D);
  gemm(stream, true, false, false, memsel, caWqkv + 512 * 256, cabqkv + 512, Vp, Bn * CK, D, D, D, D, D);
  gemm(stream, true, false, false, Qp, Kp, nullptr, csc, Qn, CK, 32, D, D, CK,
       Bn, H, (long long)Qn * D, (long long)CK * D, (long long)H * Qn * CK,
       32, 32, (long long)Qn * CK, scale);
  softmax_rows<<<Bn * H * Qn, 256, 0, stream>>>(csc, CK);
  gemm(stream, false, false, false, csc, Vp, nullptr, ao, Qn, 32, CK, CK, D, D,
       Bn, H, (long long)H * Qn * CK, (long long)CK * D, (long long)Qn * D,
       (long long)Qn * CK, 32, 32, 1.f);
  gemm(stream, true, false, false, ao, caWo, cabo, proj, BQ, D, D, D, D, D);
  residual_ln<<<BQ, 256, 0, stream>>>(qbuf, qbuf, proj, ln2w, ln2b, D);

  // ---- 5. FFN (+ln3) -> output
  gemm(stream, true, false, false, qbuf, Wff1, bff1, f1, BQ, FFn, D, D, D, FFn);
  gelu_inplace<<<(unsigned)(((size_t)BQ * FFn + 255) / 256), 256, 0, stream>>>(f1, (size_t)BQ * FFn);
  gemm(stream, true, false, false, f1, Wff2, bff2, proj, BQ, D, FFn, FFn, FFn, D);
  residual_ln<<<BQ, 256, 0, stream>>>((float*)d_out, qbuf, proj, ln3w, ln3b, D);
}
